// MIMRGNN_6433861009677
// MI455X (gfx1250) — compile-verified, matrix operand order checked by reading, not run on hardware
//
#include <hip/hip_runtime.h>
#include <hip/hip_bf16.h>

typedef __attribute__((ext_vector_type(16))) __bf16 v16bf;
typedef __attribute__((ext_vector_type(8)))  float  v8f;

#define F_MID 128
#define F_OUT 64
#define N_GRAPH 8
#define CDIV(a, b) (((a) + (b) - 1) / (b))

// ---------------- degree / normalization ----------------
__global__ void k_deg_init(float* __restrict__ deg, int n) {
    int i = blockIdx.x * blockDim.x + threadIdx.x;
    if (i < n) deg[i] = 1.0f;  // self-loop contributes 1
}

__global__ void k_deg_count(const int* __restrict__ dst, float* __restrict__ deg, int e) {
    int i = blockIdx.x * blockDim.x + threadIdx.x;
    if (i < e) atomicAdd(&deg[dst[i]], 1.0f);
}

__global__ void k_deg_rsqrt(float* __restrict__ d, int n) {
    int i = blockIdx.x * blockDim.x + threadIdx.x;
    if (i < n) d[i] = rsqrtf(d[i]);  // deg >= 1 always (self loops)
}

// ---------------- first dense transform: x (N x 2) @ W_in (2 x 128) ----------------
__global__ void k_gemm_in(const float* __restrict__ x, const float* __restrict__ W,
                          float* __restrict__ M, int n) {
    int idx = blockIdx.x * blockDim.x + threadIdx.x;
    if (idx >= n * F_MID) return;
    int i = idx >> 7, f = idx & 127;
    M[idx] = x[2 * i] * W[f] + x[2 * i + 1] * W[F_MID + f];
}

// ---------------- WMMA GEMM: M = H (nrows x 128) @ W (128 x OUTF) ----------------
// 256 threads = 8 wave32s, 64 rows per block. A and W^T staged in LDS as bf16;
// D staged back through LDS (union) for coalesced b128 global stores.
template <int OUTF>
__global__ __launch_bounds__(256) void k_gemm_wmma(const float* __restrict__ H,
                                                   const float* __restrict__ W,
                                                   float* __restrict__ M, int nrows) {
    constexpr int NCT = OUTF / 16;  // column tiles (8 or 4)
    constexpr int WPC = 8 / NCT;    // waves per column tile (1 or 2)
    constexpr int RTS = 4 / WPC;    // row tiles per wave (4 or 2)
    constexpr int DSTR = OUTF + 4;  // D tile LDS stride (floats)
    constexpr int AW_BYTES = 64 * 136 * 2 + OUTF * 136 * 2;
    constexpr int D_BYTES  = 64 * DSTR * 4;
    constexpr int SMEM_BYTES = (AW_BYTES > D_BYTES) ? AW_BYTES : D_BYTES;

    __shared__ __align__(16) char smem[SMEM_BYTES];
    auto Alds = reinterpret_cast<__bf16(*)[136]>(smem);                  // [64][136]
    auto Wlds = reinterpret_cast<__bf16(*)[136]>(smem + 64 * 136 * 2);   // [OUTF][136]
    auto Dlds = reinterpret_cast<float(*)[DSTR]>(smem);                  // [64][DSTR]

    const int tid  = threadIdx.x;
    const int row0 = blockIdx.x * 64;

    // stage W^T as bf16: float4 along n (coalesced b128 loads)
    {
        const float4* W4 = reinterpret_cast<const float4*>(W);
        for (int idx = tid; idx < 128 * (OUTF / 4); idx += 256) {
            int k = idx / (OUTF / 4), n4 = idx % (OUTF / 4);
            float4 w = W4[idx];
            Wlds[n4 * 4 + 0][k] = (__bf16)w.x;
            Wlds[n4 * 4 + 1][k] = (__bf16)w.y;
            Wlds[n4 * 4 + 2][k] = (__bf16)w.z;
            Wlds[n4 * 4 + 3][k] = (__bf16)w.w;
        }
    }
    // stage A as bf16: float4 loads
    {
        const float4* H4 = reinterpret_cast<const float4*>(H);
        for (int idx = tid; idx < 64 * 32; idx += 256) {
            int r = idx >> 5, c4 = idx & 31;
            int grow = row0 + r;
            float4 v = make_float4(0.f, 0.f, 0.f, 0.f);
            if (grow < nrows) v = H4[(long)grow * 32 + c4];
            Alds[r][c4 * 4 + 0] = (__bf16)v.x;
            Alds[r][c4 * 4 + 1] = (__bf16)v.y;
            Alds[r][c4 * 4 + 2] = (__bf16)v.z;
            Alds[r][c4 * 4 + 3] = (__bf16)v.w;
        }
    }
    __syncthreads();

    const int wave = tid >> 5;
    const int lane = tid & 31;
    const int r15  = lane & 15;
    const int hi   = lane >> 4;
    const int ct   = wave / WPC;
    const int rp   = wave % WPC;

    v8f accs[RTS];
#pragma unroll
    for (int t = 0; t < RTS; ++t) {
        const int rt = rp + t * WPC;
        v8f acc = {0.f, 0.f, 0.f, 0.f, 0.f, 0.f, 0.f, 0.f};
#pragma unroll
        for (int kk = 0; kk < 128; kk += 32) {
            v16bf a, b;
#pragma unroll
            for (int v = 0; v < 8; ++v) {
                // 16-bit A 16x32 layout: VGPR v<4 -> K=2v, v>=4 -> K=2v+8; hi-half lanes +8
                int kv = ((v < 4) ? (2 * v) : (2 * v + 8)) + hi * 8 + kk;
                a[2 * v]     = Alds[rt * 16 + r15][kv];
                a[2 * v + 1] = Alds[rt * 16 + r15][kv + 1];
                b[2 * v]     = Wlds[ct * 16 + r15][kv];
                b[2 * v + 1] = Wlds[ct * 16 + r15][kv + 1];
            }
            acc = __builtin_amdgcn_wmma_f32_16x16x32_bf16(false, a, false, b,
                                                          (short)0, acc, false, false);
        }
        accs[t] = acc;
    }
    __syncthreads();  // all LDS reads of A/W done; safe to overwrite with D

    // scatter accumulators into D tile (lanes write consecutive columns: conflict-free)
#pragma unroll
    for (int t = 0; t < RTS; ++t) {
        const int rt = rp + t * WPC;
        const int n  = ct * 16 + r15;
#pragma unroll
        for (int j = 0; j < 8; ++j)
            Dlds[rt * 16 + j + 8 * hi][n] = accs[t][j];
    }
    __syncthreads();

    // coalesced b128 global store
    for (int idx = tid; idx < 64 * (OUTF / 4); idx += 256) {
        int r = idx / (OUTF / 4), c4 = idx % (OUTF / 4);
        int grow = row0 + r;
        if (grow < nrows) {
            float4 o = make_float4(Dlds[r][c4 * 4 + 0], Dlds[r][c4 * 4 + 1],
                                   Dlds[r][c4 * 4 + 2], Dlds[r][c4 * 4 + 3]);
            reinterpret_cast<float4*>(M)[(long)grow * (OUTF / 4) + c4] = o;
        }
    }
}

// ---------------- seed output with self-loop term + bias ----------------
// H[i,:] = dinv[i]^2 * M[i,:] + b   (edge atomics add on top -> segment_sum + b)
template <int F>
__global__ void k_init_out(const float4* __restrict__ M4, const float* __restrict__ dinv,
                           const float4* __restrict__ bias4, float4* __restrict__ H4, int n) {
    constexpr int C4 = F / 4;
    int idx = blockIdx.x * blockDim.x + threadIdx.x;
    if (idx >= n * C4) return;
    int i = idx / C4, c = idx % C4;
    float di = dinv[i];
    float s = di * di;
    float4 m = M4[idx];
    float4 b = bias4[c];
    H4[idx] = make_float4(s * m.x + b.x, s * m.y + b.y, s * m.z + b.z, s * m.w + b.w);
}

// ---------------- edge scatter: one wave32 per edge ----------------
template <int F>
__global__ __launch_bounds__(256) void k_agg_edges(const int* __restrict__ src,
                                                   const int* __restrict__ dst,
                                                   const float* __restrict__ dinv,
                                                   const float* __restrict__ M,
                                                   float* __restrict__ H, int e) {
    int gid  = blockIdx.x * blockDim.x + threadIdx.x;
    int edge = gid >> 5;
    int lane = gid & 31;
    if (edge >= e) return;
    int s = src[edge], d = dst[edge];
    float nrm = dinv[s] * dinv[d];
    if constexpr (F == 128) {
        float4 mv = reinterpret_cast<const float4*>(M + (long)s * F)[lane];
        float* hp = H + (long)d * F + lane * 4;
        atomicAdd(hp + 0, nrm * mv.x);
        atomicAdd(hp + 1, nrm * mv.y);
        atomicAdd(hp + 2, nrm * mv.z);
        atomicAdd(hp + 3, nrm * mv.w);
    } else {
        float2 mv = reinterpret_cast<const float2*>(M + (long)s * F)[lane];
        float* hp = H + (long)d * F + lane * 2;
        atomicAdd(hp + 0, nrm * mv.x);
        atomicAdd(hp + 1, nrm * mv.y);
    }
}

__global__ void k_relu(float4* __restrict__ H, long n4) {
    long i = (long)blockIdx.x * blockDim.x + threadIdx.x;
    if (i < n4) {
        float4 v = H[i];
        H[i] = make_float4(fmaxf(v.x, 0.f), fmaxf(v.y, 0.f), fmaxf(v.z, 0.f), fmaxf(v.w, 0.f));
    }
}

// ---------------- pooling ----------------
__global__ void k_pool_zero(float* __restrict__ sums, float* __restrict__ cnt) {
    int i = threadIdx.x;
    if (i < N_GRAPH * F_OUT) sums[i] = 0.0f;
    if (i < N_GRAPH) cnt[i] = 0.0f;
}

__global__ __launch_bounds__(256) void k_pool_acc(const float* __restrict__ H,
                                                  const int* __restrict__ batch,
                                                  float* __restrict__ sums,
                                                  float* __restrict__ cnt, int n) {
    __shared__ float ss[N_GRAPH * F_OUT];
    __shared__ float sc[N_GRAPH];
    for (int i = threadIdx.x; i < N_GRAPH * F_OUT; i += blockDim.x) ss[i] = 0.0f;
    if (threadIdx.x < N_GRAPH) sc[threadIdx.x] = 0.0f;
    __syncthreads();

    int lane = threadIdx.x & 31;
    int warp = (blockIdx.x * blockDim.x + threadIdx.x) >> 5;
    int nw   = (gridDim.x * blockDim.x) >> 5;
    for (int node = warp; node < n; node += nw) {
        int g = batch[node];
        float2 v = reinterpret_cast<const float2*>(H + (long)node * F_OUT)[lane];
        atomicAdd(&ss[g * F_OUT + lane * 2 + 0], v.x);
        atomicAdd(&ss[g * F_OUT + lane * 2 + 1], v.y);
        if (lane == 0) atomicAdd(&sc[g], 1.0f);
    }
    __syncthreads();
    for (int i = threadIdx.x; i < N_GRAPH * F_OUT; i += blockDim.x) atomicAdd(&sums[i], ss[i]);
    if (threadIdx.x < N_GRAPH) atomicAdd(&cnt[threadIdx.x], sc[threadIdx.x]);
}

__global__ void k_finalize(const float* __restrict__ sums, const float* __restrict__ cnt,
                           float* __restrict__ out) {
    int idx = blockIdx.x * blockDim.x + threadIdx.x;
    if (idx >= N_GRAPH * F_OUT) return;
    int g = idx >> 6;
    float v = sums[idx] / fmaxf(cnt[g], 1.0f);
    out[idx] = 1.0f / (1.0f + expf(-v));
}

// ---------------- host launch ----------------
extern "C" void kernel_launch(void* const* d_in, const int* in_sizes, int n_in,
                              void* d_out, int out_size, void* d_ws, size_t ws_size,
                              hipStream_t stream) {
    const float* x     = (const float*)d_in[0];
    const int*   ei    = (const int*)d_in[1];
    const int*   batch = (const int*)d_in[2];
    const float* W_in  = (const float*)d_in[3];
    const float* b_in  = (const float*)d_in[4];
    const float* W_h   = (const float*)d_in[5];
    const float* b_h   = (const float*)d_in[6];
    const float* W_out = (const float*)d_in[7];
    const float* b_out = (const float*)d_in[8];

    const int n = in_sizes[0] / 2;  // N nodes (IN_DIM = 2)
    const int e = in_sizes[1] / 2;  // E edges
    const int* src = ei;
    const int* dst = ei + e;

    float* ws   = (float*)d_ws;
    float* dinv = ws;                          // N
    float* H    = dinv + n;                    // N x 128
    float* M    = H + (size_t)n * F_MID;       // N x 128
    float* sums = M + (size_t)n * F_MID;       // 8 x 64
    float* cnt  = sums + N_GRAPH * F_OUT;      // 8

    // degrees -> dinv
    k_deg_init<<<CDIV(n, 256), 256, 0, stream>>>(dinv, n);
    k_deg_count<<<CDIV(e, 256), 256, 0, stream>>>(dst, dinv, e);
    k_deg_rsqrt<<<CDIV(n, 256), 256, 0, stream>>>(dinv, n);

    // layer 0: x @ W_in, propagate (no relu)
    k_gemm_in<<<CDIV(n * F_MID, 256), 256, 0, stream>>>(x, W_in, M, n);
    k_init_out<128><<<CDIV(n * 32, 256), 256, 0, stream>>>(
        (const float4*)M, dinv, (const float4*)b_in, (float4*)H, n);
    k_agg_edges<128><<<CDIV(e * 32, 256), 256, 0, stream>>>(src, dst, dinv, M, H, e);

    // 4 hidden layers: WMMA GEMM, propagate, relu
    for (int l = 0; l < 4; ++l) {
        k_gemm_wmma<128><<<CDIV(n, 64), 256, 0, stream>>>(
            H, W_h + (size_t)l * F_MID * F_MID, M, n);
        k_init_out<128><<<CDIV(n * 32, 256), 256, 0, stream>>>(
            (const float4*)M, dinv, (const float4*)(b_h + l * F_MID), (float4*)H, n);
        k_agg_edges<128><<<CDIV(e * 32, 256), 256, 0, stream>>>(src, dst, dinv, M, H, e);
        k_relu<<<CDIV(n * 32, 256), 256, 0, stream>>>((float4*)H, (long)n * 32);
    }

    // output layer: 128 -> 64, propagate (no relu)
    k_gemm_wmma<64><<<CDIV(n, 64), 256, 0, stream>>>(H, W_out, M, n);
    k_init_out<64><<<CDIV(n * 16, 256), 256, 0, stream>>>(
        (const float4*)M, dinv, (const float4*)b_out, (float4*)H, n);
    k_agg_edges<64><<<CDIV(e * 32, 256), 256, 0, stream>>>(src, dst, dinv, M, H, e);

    // mean pool per graph + sigmoid
    k_pool_zero<<<1, 512, 0, stream>>>(sums, cnt);
    k_pool_acc<<<512, 256, 0, stream>>>(H, batch, sums, cnt, n);
    k_finalize<<<CDIV(N_GRAPH * F_OUT, 256), 256, 0, stream>>>(sums, cnt, (float*)d_out);
}
